// MvsformerEncoder_17351667876045
// MI455X (gfx1250) — compile-verified, matrix operand order checked by reading, not run on hardware
//
#include <hip/hip_runtime.h>
#include <hip/hip_bf16.h>

typedef __attribute__((ext_vector_type(16))) __bf16        v16bf;
typedef __attribute__((ext_vector_type(8)))  float         v8f;
typedef __attribute__((ext_vector_type(8)))  unsigned int  v8u;

#define TOK     1281
#define DIMM    100
#define CH      576
#define XW      512
#define XH      640
#define ATTN_LD 1284      // TOK padded to multiple of 4 for float4 A-loads

__device__ __forceinline__ unsigned int pack_bf16(float a, float b) {
    __bf16 x = (__bf16)a, y = (__bf16)b;
    unsigned short ux = __builtin_bit_cast(unsigned short, x);
    unsigned short uy = __builtin_bit_cast(unsigned short, y);
    return (unsigned int)ux | ((unsigned int)uy << 16);
}

// ---------------------------------------------------------------------------
// x0 init: x0[t,n] = pos_emb[t,n] + (t==0 ? cls[n] : 0). Patch GEMM atomically
// accumulates into rows 1..1280, so this fully rewrites x0 every call.
// ---------------------------------------------------------------------------
__global__ void init_x0_kernel(const float* __restrict__ pos,
                               const float* __restrict__ cls,
                               float* __restrict__ x0) {
    int i = blockIdx.x * blockDim.x + threadIdx.x;
    if (i >= TOK * DIMM) return;
    float v = pos[i];
    if (i < DIMM) v += cls[i];
    x0[i] = v;
}

// ---------------------------------------------------------------------------
// Patch-embedding GEMM (M=1280, N=100, K=147456), bf16 WMMA.
// K reordered as (ch, p1, p2) so every x access is a contiguous float2 and x
// is streamed from HBM exactly once. Block = 256 thr (8 waves): waves 0..6
// own the 7 n-tiles; wave 7 only stages. grid = (80 m-tiles, 18 ch-groups),
// partial sums atomicAdd into x0 (pre-initialized with pos_emb).
// LDS holds bf16-PAIRS (uint32) in [row][kp] order, 20-dword pitch, so the
// fragment build is two ds_load_b128 per operand.
// ---------------------------------------------------------------------------
__global__ __launch_bounds__(256)
void patch_embed_kernel(const float* __restrict__ x,
                        const float* __restrict__ Wp,
                        float* __restrict__ x0) {
    __shared__ __attribute__((aligned(16))) unsigned int sApk[16][20];  // [m][kp]
    __shared__ __attribute__((aligned(16))) unsigned int sBpk[112][20]; // [n][kp]

    const int mt     = blockIdx.x;
    const int chbase = blockIdx.y * 32;
    const int tid    = threadIdx.x;
    const int wv     = tid >> 5;
    const int lane   = tid & 31;
    const int half   = lane >> 4;
    const int idx    = lane & 15;

    // -------- thread-constant A staging coords (one float2 per chunk) --------
    const int p2h = tid & 7;            // which float2 inside the 16-float run
    const int j1  = (tid >> 3) & 1;     // p1 parity within the 32-K chunk
    const int ml  = tid >> 4;           // local m row (0..15)
    const int mg  = mt * 16 + ml;
    const int wi  = mg / 40, hi = mg % 40;              // hoisted, once
    const float* aBase = x + (size_t)(wi * 16 + j1) * XH + hi * 16 + p2h * 2;

    // -------- thread-constant B staging coords --------
    const int nB = tid & 127;           // column: <100 loads, <112 stores
    const int t7 = tid >> 7;            // kp parity (0=even,1=odd)

    v8f acc = {};

    for (int ch = chbase; ch < chbase + 32; ++ch) {
        const float* aCh = aBase + (size_t)ch * (XW * XH);
        const float* bCh = Wp + (size_t)ch * DIMM + nB;
        for (int p1g = 0; p1g < 8; ++p1g) {
            // ---- phase 1: gather to registers (loads batch under one wait) ----
            const float2 av = *(const float2*)(aCh + (size_t)p1g * (2 * XH));
            const float* bP = bCh + (size_t)p1g * 32 * ((size_t)CH * DIMM);
            float bv[16];
#pragma unroll
            for (int it = 0; it < 8; ++it) {
                const int kl0 = 2 * (t7 + 2 * it);
                if (nB < DIMM) {
                    bv[2 * it]     = bP[(size_t)kl0 * (CH * DIMM)];
                    bv[2 * it + 1] = bP[(size_t)(kl0 + 1) * (CH * DIMM)];
                } else {
                    bv[2 * it] = 0.0f; bv[2 * it + 1] = 0.0f;
                }
            }
            // ---- phase 2: pack to bf16 pairs, store to LDS ----
            sApk[ml][j1 * 8 + p2h] = pack_bf16(av.x, av.y);
            if (nB < 112) {
#pragma unroll
                for (int it = 0; it < 8; ++it)
                    sBpk[nB][t7 + 2 * it] = pack_bf16(bv[2 * it], bv[2 * it + 1]);
            }
            __syncthreads();

            if (wv < 7) {   // wave-uniform: EXEC all-1s inside WMMA waves
                const uint4* rA = (const uint4*)sApk[idx];
                const uint4* rB = (const uint4*)sBpk[wv * 16 + idx];
                const uint4 alo = rA[half],     ahi = rA[2 + half];
                const uint4 blo = rB[half],     bhi = rB[2 + half];
                const v8u au = {alo.x, alo.y, alo.z, alo.w, ahi.x, ahi.y, ahi.z, ahi.w};
                const v8u bu = {blo.x, blo.y, blo.z, blo.w, bhi.x, bhi.y, bhi.z, bhi.w};
                acc = __builtin_amdgcn_wmma_f32_16x16x32_bf16(
                    false, __builtin_bit_cast(v16bf, au),
                    false, __builtin_bit_cast(v16bf, bu),
                    (short)0, acc, false, false);
            }
            __syncthreads();
        }
    }

    const int n = wv * 16 + idx;
    if (wv < 7 && n < DIMM) {
#pragma unroll
        for (int v = 0; v < 8; ++v) {
            const int mo = mt * 16 + v + 8 * half;   // emb row; +1 skips cls
            atomicAdd(&x0[(size_t)(mo + 1) * DIMM + n], acc[v]);
        }
    }
}

// ---------------------------------------------------------------------------
// LayerNorm over DIM=100, one token per wave32, shfl_xor reductions.
// ---------------------------------------------------------------------------
__global__ __launch_bounds__(256)
void layernorm_kernel(const float* __restrict__ in,
                      const float* __restrict__ g,
                      const float* __restrict__ b,
                      float* __restrict__ out, int T) {
    const int wv = threadIdx.x >> 5, lane = threadIdx.x & 31;
    const int t  = blockIdx.x * 8 + wv;
    if (t >= T) return;
    const float* row = in + (size_t)t * DIMM;
    float v0 = row[lane];
    float v1 = row[lane + 32];
    float v2 = row[lane + 64];
    float v3 = (lane < 4) ? row[lane + 96] : 0.0f;
    float s = v0 + v1 + v2 + v3;
#pragma unroll
    for (int off = 16; off; off >>= 1) s += __shfl_xor(s, off, 32);
    const float mean = s * (1.0f / DIMM);
    float d0 = v0 - mean, d1 = v1 - mean, d2 = v2 - mean;
    float d3 = (lane < 4) ? (v3 - mean) : 0.0f;
    float q = d0 * d0 + d1 * d1 + d2 * d2 + d3 * d3;
#pragma unroll
    for (int off = 16; off; off >>= 1) q += __shfl_xor(q, off, 32);
    const float inv = rsqrtf(q * (1.0f / DIMM) + 1e-5f);
    float* o = out + (size_t)t * DIMM;
    o[lane]      = d0 * inv * g[lane]      + b[lane];
    o[lane + 32] = d1 * inv * g[lane + 32] + b[lane + 32];
    o[lane + 64] = d2 * inv * g[lane + 64] + b[lane + 64];
    if (lane < 4)
        o[lane + 96] = d3 * inv * g[lane + 96] + b[lane + 96];
}

// ---------------------------------------------------------------------------
// Generic bf16 WMMA GEMM: C = act(A@B + bias) + residual. One wave / 16x16
// tile; A loaded as 4x float4 per 32-K chunk on the fast path (lda%4==0),
// scalar fallback for K tails / M edge. Batched via grid.y (per-head attn@V).
// ---------------------------------------------------------------------------
__global__ __launch_bounds__(32)
void gemm_wmma_kernel(const float* __restrict__ A, int lda,
                      const float* __restrict__ B, int ldb,
                      float* __restrict__ C, int ldc,
                      const float* __restrict__ bias, int gelu,
                      const float* __restrict__ residual,
                      int M, int N, int K, int Mt,
                      long long aBatch, long long bBatch,
                      long long cBatch, long long rBatch) {
    const int bx = blockIdx.x, batch = blockIdx.y;
    const int mt = bx % Mt, nt = bx / Mt;
    A += (size_t)batch * aBatch;
    B += (size_t)batch * bBatch;
    C += (size_t)batch * cBatch;
    if (residual) residual += (size_t)batch * rBatch;

    const int lane = threadIdx.x;
    const int half = lane >> 4, idx = lane & 15;
    const int m_frag = mt * 16 + idx;
    const int n_frag = nt * 16 + idx;
    const bool mOK = m_frag < M;
    const bool nOK = n_frag < N;
    const bool ldaVec = (lda & 3) == 0;

    v8f acc = {};
    for (int k0 = 0; k0 < K; k0 += 32) {
        unsigned int aw[8], bw[8];
        const bool fullK = (k0 + 32) <= K;

        if (mOK && fullK && ldaVec) {
            const float* ar = A + (size_t)m_frag * lda + k0 + half * 8;
            const float4 a0 = *(const float4*)(ar);
            const float4 a1 = *(const float4*)(ar + 4);
            const float4 a2 = *(const float4*)(ar + 16);
            const float4 a3 = *(const float4*)(ar + 20);
            aw[0] = pack_bf16(a0.x, a0.y); aw[1] = pack_bf16(a0.z, a0.w);
            aw[2] = pack_bf16(a1.x, a1.y); aw[3] = pack_bf16(a1.z, a1.w);
            aw[4] = pack_bf16(a2.x, a2.y); aw[5] = pack_bf16(a2.z, a2.w);
            aw[6] = pack_bf16(a3.x, a3.y); aw[7] = pack_bf16(a3.z, a3.w);
        } else {
#pragma unroll
            for (int p = 0; p < 8; ++p) {
                const int kl = (p < 4) ? (half * 8 + 2 * p)
                                       : (16 + half * 8 + 2 * (p - 4));
                const int ka = k0 + kl, kb = ka + 1;
                const float fa = (mOK && ka < K) ? A[(size_t)m_frag * lda + ka] : 0.0f;
                const float fb = (mOK && kb < K) ? A[(size_t)m_frag * lda + kb] : 0.0f;
                aw[p] = pack_bf16(fa, fb);
            }
        }
#pragma unroll
        for (int p = 0; p < 8; ++p) {
            const int kl = (p < 4) ? (half * 8 + 2 * p)
                                   : (16 + half * 8 + 2 * (p - 4));
            const int ka = k0 + kl, kb = ka + 1;
            const float fa = (nOK && ka < K) ? B[(size_t)ka * ldb + n_frag] : 0.0f;
            const float fb = (nOK && kb < K) ? B[(size_t)kb * ldb + n_frag] : 0.0f;
            bw[p] = pack_bf16(fa, fb);
        }
        const v8u au = {aw[0], aw[1], aw[2], aw[3], aw[4], aw[5], aw[6], aw[7]};
        const v8u bu = {bw[0], bw[1], bw[2], bw[3], bw[4], bw[5], bw[6], bw[7]};
        acc = __builtin_amdgcn_wmma_f32_16x16x32_bf16(
            false, __builtin_bit_cast(v16bf, au),
            false, __builtin_bit_cast(v16bf, bu),
            (short)0, acc, false, false);
    }

    const int n = nt * 16 + idx;
    if (n >= N) return;
#pragma unroll
    for (int v = 0; v < 8; ++v) {
        const int m = mt * 16 + v + 8 * half;
        if (m >= M) continue;
        float val = acc[v];
        if (bias) val += bias[n];
        if (gelu) val = 0.5f * val * (1.0f + erff(val * 0.70710678f));
        if (residual) val += residual[(size_t)m * ldc + n];
        C[(size_t)m * ldc + n] = val;
    }
}

// ---------------------------------------------------------------------------
// scores[h,q,k] = Q[q,h*10:]·K[k,h*10:]; softmax over h (dim=1!), * 1/sqrt(100).
// 16x16 (q,k) tile per block; Q/K rows staged in LDS; attn stored with padded
// leading dim ATTN_LD so the consumer GEMM can use float4 A-loads.
// ---------------------------------------------------------------------------
__global__ __launch_bounds__(256)
void attn_scores_kernel(const float* __restrict__ Q,
                        const float* __restrict__ Km,
                        float* __restrict__ attn) {
    __shared__ float sQ[16][DIMM];
    __shared__ float sK[16][DIMM];
    const int q0 = blockIdx.x * 16, k0 = blockIdx.y * 16;
    const int r = threadIdx.x >> 4, cb = threadIdx.x & 15;
    const float* qrow = Q  + (size_t)(q0 + r) * DIMM;
    const float* krow = Km + (size_t)(k0 + r) * DIMM;
    const bool qv = (q0 + r) < TOK, kv = (k0 + r) < TOK;
#pragma unroll
    for (int j = 0; j < 7; ++j) {
        const int c = cb + j * 16;
        if (c < DIMM) {
            sQ[r][c] = qv ? qrow[c] : 0.0f;
            sK[r][c] = kv ? krow[c] : 0.0f;
        }
    }
    __syncthreads();
    const int tq = threadIdx.x >> 4, tk = threadIdx.x & 15;
    const int q = q0 + tq, k = k0 + tk;
    if (q >= TOK || k >= TOK) return;
    float s[10];
    float mx = -1e30f;
#pragma unroll
    for (int h = 0; h < 10; ++h) {
        float a = 0.0f;
#pragma unroll
        for (int d = 0; d < 10; ++d) a += sQ[tq][h * 10 + d] * sK[tk][h * 10 + d];
        s[h] = a;
        mx = fmaxf(mx, a);
    }
    float sum = 0.0f;
#pragma unroll
    for (int h = 0; h < 10; ++h) { s[h] = __expf(s[h] - mx); sum += s[h]; }
    const float inv = 0.1f / sum;                 // 1/sqrt(DIM) = 0.1
    const size_t ANN = (size_t)TOK * ATTN_LD;
#pragma unroll
    for (int h = 0; h < 10; ++h)
        attn[(size_t)h * ANN + (size_t)q * ATTN_LD + k] = s[h] * inv;
}

// ---------------------------------------------------------------------------
extern "C" void kernel_launch(void* const* d_in, const int* in_sizes, int n_in,
                              void* d_out, int out_size, void* d_ws, size_t ws_size,
                              hipStream_t stream) {
    const float* x    = (const float*)d_in[0];
    const float* Wp   = (const float*)d_in[1];
    const float* cls  = (const float*)d_in[2];
    const float* pos  = (const float*)d_in[3];
    const float* ln_g = (const float*)d_in[4];
    const float* ln_b = (const float*)d_in[5];
    const float* Wq   = (const float*)d_in[6];
    const float* Wk   = (const float*)d_in[7];
    const float* Wv   = (const float*)d_in[8];
    const float* W1   = (const float*)d_in[9];
    const float* b1   = (const float*)d_in[10];
    const float* W2   = (const float*)d_in[11];
    const float* b2   = (const float*)d_in[12];
    float* out = (float*)d_out;

    float* w = (float*)d_ws;
    const size_t T100 = (size_t)TOK * DIMM;
    float* x0   = w;
    float* x1   = w + 1 * T100;
    float* x2   = w + 2 * T100;
    float* x3   = w + 3 * T100;
    float* Q    = w + 4 * T100;
    float* Kt   = w + 5 * T100;
    float* V    = w + 6 * T100;
    float* ffh  = w + 7 * T100;                    // 1281 x 400
    float* attn = ffh + (size_t)TOK * 400;         // 10 x 1281 x ATTN_LD

    // 1) x0 = [cls; 0...] + pos_emb (rewritten every call; patch GEMM accumulates)
    init_x0_kernel<<<(TOK * DIMM + 255) / 256, 256, 0, stream>>>(pos, cls, x0);

    // 2) patch embedding: 80 m-tiles x 18 channel-group K-splits, 8 waves each
    patch_embed_kernel<<<dim3(80, 18), 256, 0, stream>>>(x, Wp, x0);

    // 3) x1 = LN(x0)
    layernorm_kernel<<<(TOK + 7) / 8, 256, 0, stream>>>(x0, ln_g, ln_b, x1, TOK);

    // 4) Q/K/V = x1 @ W{q,k,v}   (M=1281, N=100, K=100)
    gemm_wmma_kernel<<<dim3(81 * 7, 1), 32, 0, stream>>>(
        x1, DIMM, Wq, DIMM, Q, DIMM, nullptr, 0, nullptr,
        TOK, DIMM, DIMM, 81, 0, 0, 0, 0);
    gemm_wmma_kernel<<<dim3(81 * 7, 1), 32, 0, stream>>>(
        x1, DIMM, Wk, DIMM, Kt, DIMM, nullptr, 0, nullptr,
        TOK, DIMM, DIMM, 81, 0, 0, 0, 0);
    gemm_wmma_kernel<<<dim3(81 * 7, 1), 32, 0, stream>>>(
        x1, DIMM, Wv, DIMM, V, DIMM, nullptr, 0, nullptr,
        TOK, DIMM, DIMM, 81, 0, 0, 0, 0);

    // 5) scores + head-axis softmax -> attn[h][q][k] (padded rows)
    attn_scores_kernel<<<dim3(81, 81), 256, 0, stream>>>(Q, Kt, attn);

    // 6) x2 = attn @ V + x0, batched over 10 heads (M=1281, N=10, K=1281)
    gemm_wmma_kernel<<<dim3(81, 10), 32, 0, stream>>>(
        attn, ATTN_LD, V, DIMM, x2, DIMM, nullptr, 0, x0,
        TOK, 10, TOK, 81,
        (long long)TOK * ATTN_LD, 10, 10, 10);

    // 7) x3 = LN(x2)
    layernorm_kernel<<<(TOK + 7) / 8, 256, 0, stream>>>(x2, ln_g, ln_b, x3, TOK);

    // 8) ffh = gelu(x3 @ W1 + b1)   (M=1281, N=400, K=100)
    gemm_wmma_kernel<<<dim3(81 * 25, 1), 32, 0, stream>>>(
        x3, DIMM, W1, 400, ffh, 400, b1, 1, nullptr,
        TOK, 400, DIMM, 81, 0, 0, 0, 0);

    // 9) out = ffh @ W2 + b2 + x2   (M=1281, N=100, K=400)
    gemm_wmma_kernel<<<dim3(81 * 7, 1), 32, 0, stream>>>(
        ffh, 400, W2, DIMM, out, DIMM, b2, 0, x2,
        TOK, DIMM, 400, 81, 0, 0, 0, 0);

    (void)in_sizes; (void)n_in; (void)out_size; (void)ws_size;
}
